// AttentionBasedAdapter_56272661512627
// MI455X (gfx1250) — matrix operand order repaired and checked by reading, not compiled, not run on hardware
//
#include <hip/hip_runtime.h>
#include <hip/hip_bf16.h>
#include <math.h>

// Problem constants (from reference)
#define BB 16
#define TT 1024
#define DD 512
#define CC 4096
#define AA 512
#define PPD 512
#define MQ (BB * TT)   // 16384 query rows
#define CT 128         // context tile width in fused attention
#define RG 2           // row-groups of 16 per workgroup -> 32 query rows/WG

typedef __attribute__((ext_vector_type(16))) __bf16 v16bf;
typedef __attribute__((ext_vector_type(8)))  __bf16 v8bf;
typedef __attribute__((ext_vector_type(8)))  float  v8f;

#define WMMA_BF16(A_, B_, C_) \
  __builtin_amdgcn_wmma_f32_16x16x32_bf16(false, (A_), false, (B_), (short)0, (C_), false, false)

__device__ __forceinline__ v8f v8f_zero() {
  v8f z;
#pragma unroll
  for (int i = 0; i < 8; ++i) z[i] = 0.0f;
  return z;
}

// A-operand fragment (16x32 bf16): lane holds row m = lane&15.
// lanes 0-15: k = {0..15}; lanes 16-31: k = {16..31}.
// caller passes p = &row[k_chunk + (lane>=16 ? 16 : 0)]; chunks are 8 apart.
__device__ __forceinline__ v16bf make_a_frag(const __bf16* p) {
  v8bf lo = *(const v8bf*)p;
  v8bf hi = *(const v8bf*)(p + 8);
  v16bf a;
#pragma unroll
  for (int i = 0; i < 8; ++i) { a[i] = lo[i]; a[i + 8] = hi[i]; }
  return a;
}

// ---------------------------------------------------------------------------
// Weight transpose + fp32 -> bf16 cast:  W[K][N] -> WT[N][K]
// ---------------------------------------------------------------------------
__global__ void k_transpose_cast(const float* __restrict__ W,
                                 __bf16* __restrict__ WT, int K, int N) {
  int idx = blockIdx.x * 256 + threadIdx.x;  // over output elements [N][K]
  if (idx >= K * N) return;
  int n = idx / K;
  int k = idx - n * K;
  WT[idx] = (__bf16)W[(size_t)k * N + n];
}

// ---------------------------------------------------------------------------
// Y = (X @ W + b) * out_scale, bf16 output.
// X [M][K] fp32, WT [N][K] bf16 (pre-transposed), bias [N] fp32.
// Block: 256 threads = 8 waves; tile 128(m) x 64(n); wave w owns rows 16w..16w+15.
// store_transposed: Y is [N][M] instead of [M][N] (used for V -> VT).
// ---------------------------------------------------------------------------
__global__ __launch_bounds__(256) void k_gemm_bias_bf16(
    const float* __restrict__ X, const __bf16* __restrict__ WT,
    const float* __restrict__ bias, __bf16* __restrict__ Y,
    int M, int K, int N, float out_scale, int store_transposed) {
  const int lane = threadIdx.x & 31;
  const int wave = threadIdx.x >> 5;
  const int half = lane >> 4;
  const int nl   = lane & 15;
  const int m0   = blockIdx.x * 128 + wave * 16;
  const int n0   = blockIdx.y * 64;

  v8f acc[4];
#pragma unroll
  for (int nt = 0; nt < 4; ++nt) acc[nt] = v8f_zero();

  const float* xrow = X + (size_t)(m0 + nl) * K;
  for (int kk = 0; kk < K; kk += 32) {
    // A fragment: convert fp32 -> bf16 on the fly
    const float* xp = xrow + kk + half * 16;
    v16bf a;
#pragma unroll
    for (int i = 0; i < 8; ++i) {
      a[i]     = (__bf16)xp[i];
      a[i + 8] = (__bf16)xp[i + 8];
    }
#pragma unroll
    for (int nt = 0; nt < 4; ++nt) {
      const __bf16* bp = WT + (size_t)(n0 + nt * 16 + nl) * K + kk + half * 16;
      v16bf b = *(const v16bf*)bp;
      acc[nt] = WMMA_BF16(a, b, acc[nt]);
    }
  }

  // C/D layout: element (r, lane) -> m = r + 8*half, n = lane&15
#pragma unroll
  for (int nt = 0; nt < 4; ++nt) {
    const int ncol = n0 + nt * 16 + nl;
    const float bv = bias[ncol];
#pragma unroll
    for (int r = 0; r < 8; ++r) {
      const int m = m0 + r + half * 8;
      const float v = (acc[nt][r] + bv) * out_scale;
      if (store_transposed)
        Y[(size_t)ncol * M + m] = (__bf16)v;
      else
        Y[(size_t)m * N + ncol] = (__bf16)v;
    }
  }
}

// ---------------------------------------------------------------------------
// Fused flash-attention + output projection.
// Workgroup = 32 query rows (RG=2 groups of 16), 8 waves.
// Per context tile of CT=128 columns:
//   wave w: S[g] = Q_g(16xAA) . K[c0+16w ..]^T  -- every K fragment feeds RG WMMAs
//   online softmax (shfl half-wave reduce + LDS cross-wave combine, 32 rows)
//   O[g][:, 64w..64w+63] += P_g(16xCT) . V-slice -- every VT fragment feeds RG WMMAs
// Epilogue: Out = (O / l) @ Wo + bo (WoT fragments also reused RG times).
// ---------------------------------------------------------------------------
__global__ __launch_bounds__(256) void k_attn(
    const __bf16* __restrict__ Qb,   // [MQ][AA] bf16, 1/sqrt(AA) folded in
    const __bf16* __restrict__ Kb,   // [CC][AA] bf16
    const __bf16* __restrict__ VTb,  // [AA][CC] bf16
    const __bf16* __restrict__ WoT,  // [PPD][AA] bf16
    const float*  __restrict__ bo,   // [PPD] fp32
    float* __restrict__ Out)         // [MQ][PPD] fp32
{
  __shared__ __bf16 sQ[32 * AA];     // 32 KB; reused as sO in epilogue
  __shared__ __bf16 sP[32 * CT];     // 8 KB
  __shared__ float sPmax[8][32];
  __shared__ float sPsum[8][32];
  __shared__ float sMrun[32], sLrun[32], sMnew[32], sAlpha[32];

  const int tid  = threadIdx.x;
  const int lane = tid & 31;
  const int wave = tid >> 5;
  const int half = lane >> 4;
  const int nl   = lane & 15;
  const int q0   = blockIdx.x * 32;

  {  // stage the 32 query rows (bf16) into LDS: 2048 x 16B
    const v8bf* src = (const v8bf*)(Qb + (size_t)q0 * AA);
    v8bf* dst = (v8bf*)sQ;
#pragma unroll
    for (int i = 0; i < (32 * AA / 8) / 256; ++i)
      dst[tid + i * 256] = src[tid + i * 256];
  }
  if (tid < 32) { sMrun[tid] = -3.0e38f; sLrun[tid] = 0.0f; }
  __syncthreads();

  v8f o[RG][4];
#pragma unroll
  for (int g = 0; g < RG; ++g)
#pragma unroll
    for (int nt = 0; nt < 4; ++nt) o[g][nt] = v8f_zero();

  for (int c0 = 0; c0 < CC; c0 += CT) {
    const int crow = c0 + wave * 16 + nl;
    if (c0 + CT < CC)
      __builtin_prefetch(Kb + (size_t)(crow + CT) * AA, 0, 1);

    // ---- scores: each loaded K fragment feeds RG WMMAs
    v8f s[RG];
#pragma unroll
    for (int g = 0; g < RG; ++g) s[g] = v8f_zero();
    const __bf16* krow = Kb + (size_t)crow * AA + half * 16;
#pragma unroll
    for (int kk = 0; kk < AA; kk += 32) {
      v16bf b = *(const v16bf*)(krow + kk);
#pragma unroll
      for (int g = 0; g < RG; ++g) {
        v16bf a = make_a_frag(sQ + (g * 16 + nl) * AA + kk + half * 16);
        s[g] = WMMA_BF16(a, b, s[g]);
      }
    }

    // ---- per-wave row-max partials (rows split at lane 16 in C layout)
    float red[8];
#pragma unroll
    for (int g = 0; g < RG; ++g) {
#pragma unroll
      for (int r = 0; r < 8; ++r) {
        float v = s[g][r];
        v = fmaxf(v, __shfl_xor(v, 1));
        v = fmaxf(v, __shfl_xor(v, 2));
        v = fmaxf(v, __shfl_xor(v, 4));
        v = fmaxf(v, __shfl_xor(v, 8));
        red[r] = v;
      }
      if (nl == 0) {
#pragma unroll
        for (int r = 0; r < 8; ++r) sPmax[wave][g * 16 + r + half * 8] = red[r];
      }
    }
    __syncthreads();

    if (tid < 32) {  // combine across waves, update running max
      float tm = sPmax[0][tid];
#pragma unroll
      for (int w = 1; w < 8; ++w) tm = fmaxf(tm, sPmax[w][tid]);
      const float mo = sMrun[tid];
      const float mn = fmaxf(mo, tm);
      sMnew[tid]  = mn;
      sAlpha[tid] = __expf(mo - mn);
      sMrun[tid]  = mn;
    }
    __syncthreads();

    // ---- P = exp(S - mnew); stash bf16 P in LDS; partial row sums
#pragma unroll
    for (int g = 0; g < RG; ++g) {
#pragma unroll
      for (int r = 0; r < 8; ++r) {
        const int row = g * 16 + r + half * 8;
        const float p = __expf(s[g][r] - sMnew[row]);
        s[g][r] = p;
        sP[row * CT + wave * 16 + nl] = (__bf16)p;
      }
#pragma unroll
      for (int r = 0; r < 8; ++r) {
        float v = s[g][r];
        v += __shfl_xor(v, 1);
        v += __shfl_xor(v, 2);
        v += __shfl_xor(v, 4);
        v += __shfl_xor(v, 8);
        red[r] = v;
      }
      if (nl == 0) {
#pragma unroll
        for (int r = 0; r < 8; ++r) sPsum[wave][g * 16 + r + half * 8] = red[r];
      }
    }
    // rescale existing O accumulators
#pragma unroll
    for (int g = 0; g < RG; ++g)
#pragma unroll
      for (int nt = 0; nt < 4; ++nt)
#pragma unroll
        for (int r = 0; r < 8; ++r) o[g][nt][r] *= sAlpha[g * 16 + r + half * 8];
    __syncthreads();

    if (tid < 32) {  // update running denominator
      float acc = 0.0f;
#pragma unroll
      for (int w = 0; w < 8; ++w) acc += sPsum[w][tid];
      sLrun[tid] = sLrun[tid] * sAlpha[tid] + acc;
    }

    // ---- O += P @ V  (wave owns A-columns [64w, 64w+64));
    //      each loaded VT fragment feeds RG WMMAs
    const __bf16* vtbase = VTb + c0;
#pragma unroll
    for (int kc = 0; kc < CT; kc += 32) {
      v16bf a0 = make_a_frag(sP + nl * CT + kc + half * 16);
      v16bf a1 = make_a_frag(sP + (16 + nl) * CT + kc + half * 16);
#pragma unroll
      for (int nt = 0; nt < 4; ++nt) {
        const int arow = wave * 64 + nt * 16 + nl;
        v16bf b = *(const v16bf*)(vtbase + (size_t)arow * CC + kc + half * 16);
        o[0][nt] = WMMA_BF16(a0, b, o[0][nt]);
        o[1][nt] = WMMA_BF16(a1, b, o[1][nt]);
      }
    }
    __syncthreads();
  }

  // ---- normalize O and park in LDS (reuse sQ)
  __bf16* sO = sQ;
#pragma unroll
  for (int g = 0; g < RG; ++g)
#pragma unroll
    for (int nt = 0; nt < 4; ++nt)
#pragma unroll
      for (int r = 0; r < 8; ++r) {
        const int row = g * 16 + r + half * 8;
        const float v = o[g][nt][r] / sLrun[row];
        sO[row * AA + wave * 64 + nt * 16 + nl] = (__bf16)v;
      }
  __syncthreads();

  // ---- Out = Onorm @ Wo + bo; wave handles p-columns [64w, 64w+64);
  //      each WoT fragment feeds RG WMMAs
#pragma unroll
  for (int nt = 0; nt < 4; ++nt) {
    const int pcol = wave * 64 + nt * 16 + nl;
    v8f acc0 = v8f_zero();
    v8f acc1 = v8f_zero();
    const __bf16* wrow = WoT + (size_t)pcol * AA + half * 16;
#pragma unroll
    for (int kk = 0; kk < AA; kk += 32) {
      v16bf b  = *(const v16bf*)(wrow + kk);
      v16bf a0 = make_a_frag(sO + nl * AA + kk + half * 16);
      v16bf a1 = make_a_frag(sO + (16 + nl) * AA + kk + half * 16);
      acc0 = WMMA_BF16(a0, b, acc0);
      acc1 = WMMA_BF16(a1, b, acc1);
    }
    const float bv = bo[pcol];
#pragma unroll
    for (int r = 0; r < 8; ++r) {
      const int row0 = q0 + r + half * 8;
      Out[(size_t)row0 * PPD + pcol] = acc0[r] + bv;
      const int row1 = q0 + 16 + r + half * 8;
      Out[(size_t)row1 * PPD + pcol] = acc1[r] + bv;
    }
  }
}

// ---------------------------------------------------------------------------
extern "C" void kernel_launch(void* const* d_in, const int* in_sizes, int n_in,
                              void* d_out, int out_size, void* d_ws, size_t ws_size,
                              hipStream_t stream) {
  (void)in_sizes; (void)n_in; (void)out_size; (void)ws_size;
  const float* model = (const float*)d_in[0];  // [B,T,D]
  const float* ctx   = (const float*)d_in[1];  // [C,DC]
  const float* Wq = (const float*)d_in[2];
  const float* bq = (const float*)d_in[3];
  const float* Wk = (const float*)d_in[4];
  const float* bk = (const float*)d_in[5];
  const float* Wv = (const float*)d_in[6];
  const float* bv = (const float*)d_in[7];
  const float* Wo = (const float*)d_in[8];
  const float* bo = (const float*)d_in[9];
  float* out = (float*)d_out;

  char* ws = (char*)d_ws;
  size_t off = 0;
  auto alloc = [&](size_t bytes) -> void* {
    void* p = ws + off;
    off = (off + bytes + 255) & ~(size_t)255;
    return p;
  };
  __bf16* qb  = (__bf16*)alloc((size_t)MQ * AA * 2);   // Q bf16 (scale folded)
  __bf16* kb  = (__bf16*)alloc((size_t)CC * AA * 2);   // K bf16 row-major
  __bf16* vtb = (__bf16*)alloc((size_t)AA * CC * 2);   // V^T bf16
  __bf16* wqT = (__bf16*)alloc((size_t)AA * DD * 2);
  __bf16* wkT = (__bf16*)alloc((size_t)AA * DD * 2);
  __bf16* wvT = (__bf16*)alloc((size_t)AA * DD * 2);
  __bf16* woT = (__bf16*)alloc((size_t)PPD * AA * 2);

  {
    dim3 g((DD * AA + 255) / 256);
    k_transpose_cast<<<g, 256, 0, stream>>>(Wq, wqT, DD, AA);
    k_transpose_cast<<<g, 256, 0, stream>>>(Wk, wkT, DD, AA);
    k_transpose_cast<<<g, 256, 0, stream>>>(Wv, wvT, DD, AA);
    k_transpose_cast<<<g, 256, 0, stream>>>(Wo, woT, AA, PPD);
  }

  const float qscale = 1.0f / sqrtf((float)AA);
  k_gemm_bias_bf16<<<dim3(MQ / 128, AA / 64), 256, 0, stream>>>(
      model, wqT, bq, qb, MQ, DD, AA, qscale, 0);
  k_gemm_bias_bf16<<<dim3(CC / 128, AA / 64), 256, 0, stream>>>(
      ctx, wkT, bk, kb, CC, DD, AA, 1.0f, 0);
  k_gemm_bias_bf16<<<dim3(CC / 128, AA / 64), 256, 0, stream>>>(
      ctx, wvT, bv, vtb, CC, DD, AA, 1.0f, 1);  // transposed store -> VT[AA][CC]

  k_attn<<<dim3(MQ / 32), 256, 0, stream>>>(qb, kb, vtb, woT, bo, out);
}
